// SelfAttention_89361089561500
// MI455X (gfx1250) — matrix operand order repaired and checked by reading, not compiled, not run on hardware
//
#include <hip/hip_runtime.h>
#include <math.h>

// ---------------------------------------------------------------- types
typedef __attribute__((ext_vector_type(16))) __bf16 v16bf;
typedef __attribute__((ext_vector_type(8)))  __bf16 v8bf;
typedef __attribute__((ext_vector_type(4)))  __bf16 v4bf;
typedef __attribute__((ext_vector_type(8)))  float  v8f;

#define B_  8
#define T_  2048
#define C_  1024
// scale = CONTEXT_SIZE**-0.5 = 2048**-0.5
#define SCALE_ 0.02209708691207961f

// ---------------------------------------------------------------- async global->LDS (CDNA5)
// Per-lane: LDS[lds_addr] = MEM[gaddr] for 16 bytes; tracked by ASYNCcnt.
// Low 32 bits of a generic LDS pointer are the wave-relative LDS address.
__device__ __forceinline__ void async_copy16B(const __bf16* gsrc, __bf16* lds_dst) {
  unsigned lds = (unsigned)(size_t)(void*)lds_dst;
  asm volatile("global_load_async_to_lds_b128 %0, %1, off"
               :: "v"(lds), "v"(gsrc) : "memory");
}
__device__ __forceinline__ void wait_async_0() {
  asm volatile("s_wait_asynccnt 0" ::: "memory");
}
__device__ __forceinline__ void wait_async_4() {
  asm volatile("s_wait_asynccnt 4" ::: "memory");
}
__device__ __forceinline__ void wait_async_6() {
  asm volatile("s_wait_asynccnt 6" ::: "memory");
}

// ---------------------------------------------------------------- WMMA fragment loaders
// LDS tiles: A = [256 rows][32 k], B(t) = [128 cols][32 k], bf16 row-major.
// A-matrix 16x32 bf16 layout: lanes 0-15 -> M=lane, elems K=0..7,16..23;
//                             lanes 16-31 -> M=lane-16, elems K=8..15,24..31.
__device__ __forceinline__ v16bf frag_a(const __bf16* tile, int lane, int rbase) {
  int row = rbase + (lane & 15);
  int kb  = (lane >> 4) << 3;           // 0 or 8
  const __bf16* p = tile + row * 32 + kb;
  v8bf lo = *(const v8bf*)p;            // K kb..kb+7
  v8bf hi = *(const v8bf*)(p + 16);     // K kb+16..kb+23
  return __builtin_shufflevector(lo, hi, 0,1,2,3,4,5,6,7,8,9,10,11,12,13,14,15);
}
// B-matrix 32x16 bf16 layout: lane holds column N=lane&15;
// lanes 0-15 hold K=0..7,16..23, lanes 16-31 hold K=8..15,24..31 (matches A).
__device__ __forceinline__ v16bf frag_b(const __bf16* tile, int lane, int nbase) {
  int n  = nbase + (lane & 15);
  int kb = (lane >> 4) << 3;            // 0 or 8
  const __bf16* p = tile + n * 32 + kb;
  v8bf lo = *(const v8bf*)p;
  v8bf hi = *(const v8bf*)(p + 16);
  return __builtin_shufflevector(lo, hi, 0,1,2,3,4,5,6,7,8,9,10,11,12,13,14,15);
}

// ---------------------------------------------------------------- staging
// BT_COPY=true : B source rows already ARE the needed columns (Q.K^T), async copy.
// BT_COPY=false: B source row-major [K][N], manual transpose into Bt[n][k] (+prefetch).
template <bool BT_COPY>
__device__ __forceinline__ void stage(const __bf16* __restrict__ A, int lda,
                                      const __bf16* __restrict__ Bm, int ldb,
                                      int m0, int n0, int k0,
                                      __bf16* As, __bf16* Bs) {
  const int tid = threadIdx.x;
  // A tile 256x32: one row per thread, 4 async 16B chunks -> 4 async ops/wave
  {
    const __bf16* g = A + (size_t)(m0 + tid) * lda + k0;
    __bf16* l = As + tid * 32;
#pragma unroll
    for (int c = 0; c < 4; ++c) async_copy16B(g + c * 8, l + c * 8);
  }
  if (BT_COPY) {  // 2 async ops/wave
    int n = tid >> 1, kc = (tid & 1) * 16;
    const __bf16* g = Bm + (size_t)(n0 + n) * ldb + k0 + kc;
    __bf16* l = Bs + n * 32 + kc;
    async_copy16B(g, l);
    async_copy16B(g + 8, l + 8);
  } else {
    int kk = tid >> 3, nb = (tid & 7) * 16;
    const __bf16* g = Bm + (size_t)(k0 + kk) * ldb + n0 + nb;
    __builtin_prefetch(g + 32 * (size_t)ldb, 0, 1);  // next k-slab
    v8bf b0 = *(const v8bf*)g;
    v8bf b1 = *(const v8bf*)(g + 8);
#pragma unroll
    for (int i = 0; i < 8; ++i) Bs[(nb + i) * 32 + kk] = b0[i];
#pragma unroll
    for (int i = 0; i < 8; ++i) Bs[(nb + 8 + i) * 32 + kk] = b1[i];
  }
}

// ---------------------------------------------------------------- GEMM mainloop
// Block tile 256x128, 256 threads = 8 waves in 4(M)x2(N), wave tile 64x64
// (4x4 16x16x32 WMMA tiles), K-step 32, double-buffered LDS with async fills.
template <bool BT_COPY>
__device__ __forceinline__ void gemm_core(const __bf16* __restrict__ A, int lda,
                                          const __bf16* __restrict__ Bm, int ldb,
                                          int m0, int n0, int kend,
                                          __bf16* As0, __bf16* Bs0,
                                          __bf16* As1, __bf16* Bs1,
                                          v8f acc[4][4]) {
  const int lane = threadIdx.x & 31;
  const int wave = threadIdx.x >> 5;
  const int wm   = (wave & 3) * 64;
  const int wn   = (wave >> 2) * 64;

  stage<BT_COPY>(A, lda, Bm, ldb, m0, n0, 0, As0, Bs0);
  for (int k0 = 0; k0 < kend; k0 += 32) {
    __bf16* Ac = (k0 & 32) ? As1 : As0;
    __bf16* Bc = (k0 & 32) ? Bs1 : Bs0;
    __bf16* An = (k0 & 32) ? As0 : As1;
    __bf16* Bn = (k0 & 32) ? Bs0 : Bs1;
    const bool has_next = (k0 + 32) < kend;
    if (has_next) stage<BT_COPY>(A, lda, Bm, ldb, m0, n0, k0 + 32, An, Bn);
    // wait for *current* buffer's async fills (newest ops belong to next buffer)
    if (has_next) { if (BT_COPY) wait_async_6(); else wait_async_4(); }
    else          wait_async_0();
    __syncthreads();

    v16bf af[4];
#pragma unroll
    for (int mt = 0; mt < 4; ++mt) af[mt] = frag_a(Ac, lane, wm + mt * 16);
#pragma unroll
    for (int nt = 0; nt < 4; ++nt) {
      v16bf bfr = frag_b(Bc, lane, wn + nt * 16);
#pragma unroll
      for (int mt = 0; mt < 4; ++mt)
        acc[mt][nt] = __builtin_amdgcn_wmma_f32_16x16x32_bf16(
            false, af[mt], false, bfr, (short)0, acc[mt][nt], false, false);
    }
    __syncthreads();
  }
}

// ---------------------------------------------------------------- cast f32 -> bf16
__global__ __launch_bounds__(256) void cast_f32_to_bf16(const float* __restrict__ src,
                                                        __bf16* __restrict__ dst, int n4) {
  int i = blockIdx.x * blockDim.x + threadIdx.x;
  if (i >= n4) return;
  float4 f = ((const float4*)src)[i];
  v4bf o;
  o[0] = (__bf16)f.x; o[1] = (__bf16)f.y; o[2] = (__bf16)f.z; o[3] = (__bf16)f.w;
  ((v4bf*)dst)[i] = o;
}

// ---------------------------------------------------------------- kernel 1: QKV projection
__global__ __launch_bounds__(256) void proj_gemm(const __bf16* __restrict__ X,
                                                 const __bf16* __restrict__ W,
                                                 const float* __restrict__ bias,
                                                 __bf16* __restrict__ Out) {
  __shared__ __bf16 As[2][256 * 32];
  __shared__ __bf16 Bs[2][128 * 32];
  const int m0 = blockIdx.x * 256, n0 = blockIdx.y * 128;
  const int lane = threadIdx.x & 31, wave = threadIdx.x >> 5;
  const int wm = (wave & 3) * 64, wn = (wave >> 2) * 64;
  v8f acc[4][4] = {};
  gemm_core<false>(X, C_, W, C_, m0, n0, C_, As[0], Bs[0], As[1], Bs[1], acc);
#pragma unroll
  for (int mt = 0; mt < 4; ++mt)
#pragma unroll
    for (int nt = 0; nt < 4; ++nt) {
      int col   = n0 + wn + nt * 16 + (lane & 15);
      float bv  = bias[col];
      int rbase = m0 + wm + mt * 16 + ((lane >> 4) << 3);
#pragma unroll
      for (int j = 0; j < 8; ++j)
        Out[(size_t)(rbase + j) * C_ + col] = (__bf16)(acc[mt][nt][j] + bv);
    }
}

// ---------------------------------------------------------------- kernel 2: causal scores
__global__ __launch_bounds__(256) void attn_scores(const __bf16* __restrict__ Q,
                                                   const __bf16* __restrict__ Kb,
                                                   float* __restrict__ S) {
  const int m0 = blockIdx.x * 256, n0 = blockIdx.y * 128, bz = blockIdx.z;
  if (n0 > m0 + 255) return;  // fully above diagonal: softmax never reads it
  __shared__ __bf16 As[2][256 * 32];
  __shared__ __bf16 Bs[2][128 * 32];
  const int lane = threadIdx.x & 31, wave = threadIdx.x >> 5;
  const int wm = (wave & 3) * 64, wn = (wave >> 2) * 64;
  const __bf16* A  = Q  + (size_t)bz * T_ * C_;
  const __bf16* Bm = Kb + (size_t)bz * T_ * C_;
  v8f acc[4][4] = {};
  gemm_core<true>(A, C_, Bm, C_, m0, n0, C_, As[0], Bs[0], As[1], Bs[1], acc);
  float* Sb = S + (size_t)bz * T_ * T_;
#pragma unroll
  for (int mt = 0; mt < 4; ++mt)
#pragma unroll
    for (int nt = 0; nt < 4; ++nt) {
      int col   = n0 + wn + nt * 16 + (lane & 15);
      int rbase = m0 + wm + mt * 16 + ((lane >> 4) << 3);
#pragma unroll
      for (int j = 0; j < 8; ++j) {
        int row = rbase + j;
        float v = acc[mt][nt][j] * SCALE_;
        if (col > row) v = -__builtin_inff();
        Sb[(size_t)row * T_ + col] = v;
      }
    }
}

// ---------------------------------------------------------------- kernel 3: row softmax -> P bf16
// Masked cols (> tq) forced to -inf on read; written as exact zeros so the PV
// GEMM can consume full tiles blindly.
__global__ __launch_bounds__(256) void row_softmax(const float* __restrict__ S,
                                                   __bf16* __restrict__ P) {
  __shared__ float red[256];
  const int row = blockIdx.x;  // 0 .. B*T-1
  const int tq  = row % T_;
  const int tid = threadIdx.x;
  const float* Srow = S + (size_t)row * T_;
  __bf16* Prow      = P + (size_t)row * T_;

  float v[8];
  const int c0 = tid * 8;
  float4 f0 = ((const float4*)(Srow + c0))[0];
  float4 f1 = ((const float4*)(Srow + c0))[1];
  v[0] = f0.x; v[1] = f0.y; v[2] = f0.z; v[3] = f0.w;
  v[4] = f1.x; v[5] = f1.y; v[6] = f1.z; v[7] = f1.w;
#pragma unroll
  for (int i = 0; i < 8; ++i)
    if (c0 + i > tq) v[i] = -__builtin_inff();

  float mx = v[0];
#pragma unroll
  for (int i = 1; i < 8; ++i) mx = fmaxf(mx, v[i]);
  red[tid] = mx;
  __syncthreads();
  for (int s = 128; s > 0; s >>= 1) {
    if (tid < s) red[tid] = fmaxf(red[tid], red[tid + s]);
    __syncthreads();
  }
  mx = red[0];
  __syncthreads();

  float e[8], sum = 0.f;
#pragma unroll
  for (int i = 0; i < 8; ++i) { e[i] = __expf(v[i] - mx); sum += e[i]; }
  red[tid] = sum;
  __syncthreads();
  for (int s = 128; s > 0; s >>= 1) {
    if (tid < s) red[tid] += red[tid + s];
    __syncthreads();
  }
  const float inv = 1.0f / red[0];

  v8bf o;
#pragma unroll
  for (int i = 0; i < 8; ++i) o[i] = (__bf16)(e[i] * inv);
  *(v8bf*)(Prow + c0) = o;
}

// ---------------------------------------------------------------- kernel 4: O = P @ V
__global__ __launch_bounds__(256) void attn_out(const __bf16* __restrict__ P,
                                                const __bf16* __restrict__ V,
                                                float* __restrict__ O) {
  __shared__ __bf16 As[2][256 * 32];
  __shared__ __bf16 Bs[2][128 * 32];
  const int m0 = blockIdx.x * 256, n0 = blockIdx.y * 128, bz = blockIdx.z;
  const int lane = threadIdx.x & 31, wave = threadIdx.x >> 5;
  const int wm = (wave & 3) * 64, wn = (wave >> 2) * 64;
  const __bf16* A  = P + (size_t)bz * T_ * T_;
  const __bf16* Bm = V + (size_t)bz * T_ * C_;
  const int kend = m0 + 256;  // causal truncation (exact: P==0 beyond diagonal)
  v8f acc[4][4] = {};
  gemm_core<false>(A, T_, Bm, C_, m0, n0, kend, As[0], Bs[0], As[1], Bs[1], acc);
  float* Ob = O + (size_t)bz * T_ * C_;
#pragma unroll
  for (int mt = 0; mt < 4; ++mt)
#pragma unroll
    for (int nt = 0; nt < 4; ++nt) {
      int col   = n0 + wn + nt * 16 + (lane & 15);
      int rbase = m0 + wm + mt * 16 + ((lane >> 4) << 3);
#pragma unroll
      for (int j = 0; j < 8; ++j)
        Ob[(size_t)(rbase + j) * C_ + col] = acc[mt][nt][j];
    }
}

// ---------------------------------------------------------------- launch
extern "C" void kernel_launch(void* const* d_in, const int* in_sizes, int n_in,
                              void* d_out, int out_size, void* d_ws, size_t ws_size,
                              hipStream_t stream) {
  const float* x  = (const float*)d_in[0];
  const float* Wq = (const float*)d_in[1];
  const float* bq = (const float*)d_in[2];
  const float* Wk = (const float*)d_in[3];
  const float* bk = (const float*)d_in[4];
  const float* Wv = (const float*)d_in[5];
  const float* bv = (const float*)d_in[6];
  float* out = (float*)d_out;

  const size_t NTC = (size_t)B_ * T_ * C_;  // 16,777,216
  const size_t NCC = (size_t)C_ * C_;       // 1,048,576

  __bf16* x_bf  = (__bf16*)d_ws;
  __bf16* wq_bf = x_bf + NTC;
  __bf16* wk_bf = wq_bf + NCC;
  __bf16* wv_bf = wk_bf + NCC;
  __bf16* q_bf  = wv_bf + NCC;
  __bf16* k_bf  = q_bf + NTC;
  __bf16* v_bf  = k_bf + NTC;
  float*  Smat  = (float*)(v_bf + NTC);
  __bf16* Pmat  = q_bf;  // alias: q+k (2*NTC bf16 == B*T*T bf16), dead once S is built

  // 1) casts
  cast_f32_to_bf16<<<(int)(NTC / 4 / 256), 256, 0, stream>>>(x, x_bf, (int)(NTC / 4));
  cast_f32_to_bf16<<<(int)(NCC / 4 / 256), 256, 0, stream>>>(Wq, wq_bf, (int)(NCC / 4));
  cast_f32_to_bf16<<<(int)(NCC / 4 / 256), 256, 0, stream>>>(Wk, wk_bf, (int)(NCC / 4));
  cast_f32_to_bf16<<<(int)(NCC / 4 / 256), 256, 0, stream>>>(Wv, wv_bf, (int)(NCC / 4));

  // 2) Q/K/V projections: [B*T, C] x [C, C]
  dim3 gproj(B_ * T_ / 256, C_ / 128);
  proj_gemm<<<gproj, 256, 0, stream>>>(x_bf, wq_bf, bq, q_bf);
  proj_gemm<<<gproj, 256, 0, stream>>>(x_bf, wk_bf, bk, k_bf);
  proj_gemm<<<gproj, 256, 0, stream>>>(x_bf, wv_bf, bv, v_bf);

  // 3) causal scores
  dim3 gsc(T_ / 256, T_ / 128, B_);
  attn_scores<<<gsc, 256, 0, stream>>>(q_bf, k_bf, Smat);

  // 4) softmax rows -> P bf16 (overwrites q/k region)
  row_softmax<<<B_ * T_, 256, 0, stream>>>(Smat, Pmat);

  // 5) O = P @ V
  dim3 gout(T_ / 256, C_ / 128, B_);
  attn_out<<<gout, 256, 0, stream>>>(Pmat, v_bf, out);
}
